// MinkTrunk_64707977281697
// MI455X (gfx1250) — matrix operand order repaired and checked by reading, not compiled, not run on hardware
//
#include <hip/hip_runtime.h>

// ---------------------------------------------------------------------------
// MinkowskiNet trunk on gfx1250.
// Sparse convs: gather(bf16) -> v_wmma_f32_16x16x32_bf16 (2x2 macro-tile)
//               -> f32 atomic scatter-add.
// Activations converted to bf16 once per layer (fused in BN/residual kernels);
// weights converted + transposed to bf16 [K][Cout][Cin] once per launch.
// ---------------------------------------------------------------------------

typedef __bf16 bf16_t;
typedef bf16_t v16bf __attribute__((ext_vector_type(16)));
typedef bf16_t v8bf  __attribute__((ext_vector_type(8)));
typedef float  v8f   __attribute__((ext_vector_type(8)));

#define BN_EPS 1e-5f

// round-to-nearest-even f32 -> bf16 (bit-exact, no reliance on cast lowering)
__device__ __forceinline__ bf16_t f2bf(float f) {
  unsigned u = __builtin_bit_cast(unsigned, f);
  u += 0x7fffu + ((u >> 16) & 1u);
  unsigned short h = (unsigned short)(u >> 16);
  return __builtin_bit_cast(bf16_t, h);
}
__device__ __forceinline__ bf16_t bf_zero() {
  return __builtin_bit_cast(bf16_t, (unsigned short)0);
}

// two 16B chunks -> one 16-element bf16 fragment register group
__device__ __forceinline__ v16bf ld16(const bf16_t* p0, const bf16_t* p1) {
  v8bf a = *(const v8bf*)p0;
  v8bf b = *(const v8bf*)p1;
  return __builtin_shufflevector(a, b, 0, 1, 2, 3, 4, 5, 6, 7,
                                 8, 9, 10, 11, 12, 13, 14, 15);
}

#define WMMA_BF16(A, B, C)                                                    \
  __builtin_amdgcn_wmma_f32_16x16x32_bf16(false, (A), false, (B), (short)0,   \
                                          (C), false, false)

// ---------------------------------------------------------------------------
__global__ void k_zero(float* __restrict__ p, long n) {
  long i = (long)blockIdx.x * blockDim.x + threadIdx.x;
  long s = (long)gridDim.x * blockDim.x;
  for (; i < n; i += s) p[i] = 0.f;
}

// W (K,Cin,Cout) f32  ->  Wt (K,Cout,Cin) bf16  (runs once per launch, small)
__global__ void k_w2bf(const float* __restrict__ W, bf16_t* __restrict__ Wt,
                       int K, int Cin, int Cout) {
  long total = (long)K * Cin * Cout;
  long i = (long)blockIdx.x * blockDim.x + threadIdx.x;
  long s = (long)gridDim.x * blockDim.x;
  for (; i < total; i += s) {
    int k  = (int)(i / ((long)Cin * Cout));
    int r  = (int)(i % ((long)Cin * Cout));
    int ci = r / Cout;
    int co = r % Cout;
    Wt[((size_t)k * Cout + co) * Cin + ci] = f2bf(W[i]);
  }
}

// Stem conv: Cin = 1, Cout = 64, scalar path (WMMA is waste at K-dim = 1).
__global__ void k_stem(const float* __restrict__ feats, const float* __restrict__ W,
                       const int* __restrict__ iidx, const int* __restrict__ oidx,
                       float* __restrict__ out, int n0, int P) {
  int k = blockIdx.y;
  int p = blockIdx.x * (blockDim.x >> 6) + (threadIdx.x >> 6);
  int c = threadIdx.x & 63;
  if (p >= P) return;
  int o = oidx[(size_t)k * P + p];
  if (o == n0) return;                     // padded pair
  int ii = iidx[(size_t)k * P + p];
  atomicAdd(&out[(size_t)o * 64 + c], feats[ii] * W[k * 64 + c]);
}

// ---------------------------------------------------------------------------
// Sparse conv, 2x2 macro-tile: one wave covers 32 pairs x 32 couts per offset.
// xb: (n_in+1) x Cin bf16, row n_in zero.  Wt: (K,Cout,Cin) bf16.
// out: (n_out+1) x Cout f32, pre-zeroed; scatter via f32 atomics.
__global__ void k_conv_wmma(const bf16_t* __restrict__ xb, const bf16_t* __restrict__ Wt,
                            const int* __restrict__ iidx, const int* __restrict__ oidx,
                            float* __restrict__ out,
                            int P, int Cin, int Cout, int n_in, int n_out) {
  int lane = threadIdx.x;                                  // 0..31 (wave32)
  int ct   = blockIdx.y * blockDim.y + threadIdx.y;        // 32-col tile
  if (ct * 32 >= Cout) return;
  int cb = ct * 32;
  int k  = blockIdx.z;
  int m  = lane & 15;
  int pb = blockIdx.x * 32;
  int pa0 = pb + m, pa1 = pb + 16 + m;

  int or0 = (pa0 < P) ? oidx[(size_t)k * P + pa0] : n_out;
  int or1 = (pa1 < P) ? oidx[(size_t)k * P + pa1] : n_out;
  if (__ballot((or0 != n_out) || (or1 != n_out)) == 0ull) return;
  int ir0 = (pa0 < P) ? iidx[(size_t)k * P + pa0] : n_in;
  int ir1 = (pa1 < P) ? iidx[(size_t)k * P + pa1] : n_in;

  int lo  = (lane < 16) ? 0 : 8;                           // A-fragment K base
  int Klo = (lane < 16) ? 0 : 16;                          // unused: B loads also use lo
  const bf16_t* x0 = xb + (size_t)ir0 * Cin;
  const bf16_t* x1 = xb + (size_t)ir1 * Cin;
  const bf16_t* w0 = Wt + ((size_t)k * Cout + cb + m) * Cin;   // col cb+m
  const bf16_t* w1 = w0 + (size_t)16 * Cin;                    // col cb+16+m

  v8f a00 = {}, a01 = {}, a10 = {}, a11 = {};
  for (int kc = 0; kc < Cin; kc += 32) {
    v16bf A0 = ld16(x0 + kc + lo, x0 + kc + 16 + lo);
    v16bf A1 = ld16(x1 + kc + lo, x1 + kc + 16 + lo);
    v16bf B0 = ld16(w0 + kc + lo, w0 + kc + 16 + lo);
    v16bf B1 = ld16(w1 + kc + lo, w1 + kc + 16 + lo);
    a00 = WMMA_BF16(A0, B0, a00);
    a01 = WMMA_BF16(A0, B1, a01);
    a10 = WMMA_BF16(A1, B0, a10);
    a11 = WMMA_BF16(A1, B1, a11);
  }

  int mh   = (lane >> 4) << 3;                             // C/D: VGPR r -> M = mh+r
  int col0 = cb + m, col1 = cb + 16 + m;
#pragma unroll
  for (int r = 0; r < 8; ++r) {
    int od0 = __shfl(or0, mh + r, 32);
    if (od0 != n_out) {
      atomicAdd(&out[(size_t)od0 * Cout + col0], a00[r]);
      atomicAdd(&out[(size_t)od0 * Cout + col1], a01[r]);
    }
    int od1 = __shfl(or1, mh + r, 32);
    if (od1 != n_out) {
      atomicAdd(&out[(size_t)od1 * Cout + col0], a10[r]);
      atomicAdd(&out[(size_t)od1 * Cout + col1], a11[r]);
    }
  }
}

// Dense 1x1 conv: out(n,Cout) = xb(n,Cin) @ Wt(Cout,Cin)^T ; plain stores.
__global__ void k_gemm_wmma(const bf16_t* __restrict__ xb, const bf16_t* __restrict__ Wt,
                            float* __restrict__ out, int n, int Cin, int Cout) {
  int lane = threadIdx.x;
  int ct   = blockIdx.y * blockDim.y + threadIdx.y;
  if (ct * 16 >= Cout) return;
  int cb  = ct * 16;
  int m   = lane & 15;
  int row = blockIdx.x * 16 + m;
  int irw = (row < n) ? row : n;                           // zero pad row
  int lo  = (lane < 16) ? 0 : 8;
  int Klo = (lane < 16) ? 0 : 16;
  const bf16_t* xr = xb + (size_t)irw * Cin;
  const bf16_t* wr = Wt + (size_t)(cb + m) * Cin;

  v8f acc = {};
  for (int kc = 0; kc < Cin; kc += 32) {
    v16bf A = ld16(xr + kc + lo, xr + kc + 16 + lo);
    v16bf B = ld16(wr + kc + lo, wr + kc + 16 + lo);
    acc = WMMA_BF16(A, B, acc);
  }

  int mh  = (lane >> 4) << 3;
  int col = cb + m;
#pragma unroll
  for (int r = 0; r < 8; ++r) {
    int orw = blockIdx.x * 16 + mh + r;
    if (orw < n) out[(size_t)orw * Cout + col] = acc[r];
  }
}

// ---------------------------------------------------------------------------
// BN (training mode, biased variance) over the point dimension.
__global__ void k_bn_reduce(const float* __restrict__ x, float* __restrict__ stats,
                            int n, int C, int rowsPerBlock) {
  int c = threadIdx.x;                 // blockDim.x == C
  long r0 = (long)blockIdx.x * rowsPerBlock;
  long r1 = r0 + rowsPerBlock; if (r1 > n) r1 = n;
  float s = 0.f, q = 0.f;
  for (long r = r0; r < r1; ++r) {
    float v = x[r * C + c];
    s += v; q += v * v;
  }
  atomicAdd(&stats[c], s);
  atomicAdd(&stats[C + c], q);
}

// In-place BN(+ReLU) on f32; optionally mirrors result to bf16 (with zero pad
// row n) for the next conv's gather input.
__global__ void k_bn_apply(float* __restrict__ x, const float* __restrict__ stats,
                           const float* __restrict__ g, const float* __restrict__ b,
                           bf16_t* __restrict__ xb, int n, int C, int relu) {
  long i = (long)blockIdx.x * blockDim.x + threadIdx.x;
  if (i >= (long)(n + 1) * C) return;
  if (i >= (long)n * C) {                  // pad row
    if (xb) xb[i] = bf_zero();
    return;
  }
  int c = (int)(i % C);
  float mean = stats[c] / (float)n;
  float var  = stats[C + c] / (float)n - mean * mean;
  float v = (x[i] - mean) * rsqrtf(var + BN_EPS) * g[c] + b[c];
  if (relu) v = fmaxf(v, 0.f);
  x[i] = v;
  if (xb) xb[i] = f2bf(v);
}

// y = relu(h + res) -> d_out section (f32) and optional bf16 next-conv input.
__global__ void k_add_relu(const float* __restrict__ h, const float* __restrict__ res,
                           bf16_t* __restrict__ yb, float* __restrict__ dout,
                           int n, int C) {
  long i = (long)blockIdx.x * blockDim.x + threadIdx.x;
  if (i >= (long)(n + 1) * C) return;
  if (i >= (long)n * C) {                  // pad row
    if (yb) yb[i] = bf_zero();
    return;
  }
  float v = fmaxf(h[i] + res[i], 0.f);
  if (yb) yb[i] = f2bf(v);
  dout[i] = v;
}

// ---------------------------------------------------------------------------
static void run_bn(float* x, float* stats, const float* g, const float* b,
                   bf16_t* xb, int n, int C, int relu, hipStream_t s) {
  k_zero<<<1, 512, 0, s>>>(stats, 2 * C);
  const int rows = 128;
  k_bn_reduce<<<dim3((n + rows - 1) / rows), dim3(C), 0, s>>>(x, stats, n, C, rows);
  long total = (long)(n + 1) * C;
  k_bn_apply<<<dim3((unsigned)((total + 255) / 256)), 256, 0, s>>>(x, stats, g, b, xb, n, C, relu);
}

static void run_conv(const bf16_t* xb, const bf16_t* Wt, const int* ii, const int* oi,
                     float* out, int K, int P, int Cin, int Cout,
                     int n_in, int n_out, hipStream_t s) {
  k_zero<<<dim3(1024), 256, 0, s>>>(out, (long)(n_out + 1) * Cout);
  dim3 blk(32, 2);
  dim3 grd((P + 31) / 32, (Cout / 32 + 1) / 2, K);
  k_conv_wmma<<<grd, blk, 0, s>>>(xb, Wt, ii, oi, out, P, Cin, Cout, n_in, n_out);
}

static void run_gemm(const bf16_t* xb, const bf16_t* Wt, float* out,
                     int n, int Cin, int Cout, hipStream_t s) {
  dim3 blk(32, 2);
  dim3 grd((n + 15) / 16, (Cout / 16 + 1) / 2, 1);
  k_gemm_wmma<<<grd, blk, 0, s>>>(xb, Wt, out, n, Cin, Cout);
}

static bf16_t* xpose(const float* W, bf16_t* dst, int K, int Cin, int Cout,
                     bf16_t** cur, hipStream_t s) {
  (void)dst;
  bf16_t* p = *cur;
  long total = (long)K * Cin * Cout;
  k_w2bf<<<dim3((unsigned)((total + 255) / 256)), 256, 0, s>>>(W, p, K, Cin, Cout);
  *cur = p + total;
  return p;
}

extern "C" void kernel_launch(void* const* d_in, const int* in_sizes, int n_in,
                              void* d_out, int out_size, void* d_ws, size_t ws_size,
                              hipStream_t stream) {
  (void)n_in; (void)out_size; (void)ws_size;
  const float* feats = (const float*)d_in[0];
  const float* W0  = (const float*)d_in[1];  const float* g0  = (const float*)d_in[2];  const float* b0  = (const float*)d_in[3];
  const float* Ws1 = (const float*)d_in[4];  const float* gs1 = (const float*)d_in[5];  const float* bs1 = (const float*)d_in[6];
  const float* Wa1 = (const float*)d_in[7];  const float* ga1 = (const float*)d_in[8];  const float* ba1 = (const float*)d_in[9];
  const float* Wb1 = (const float*)d_in[10]; const float* gb1 = (const float*)d_in[11]; const float* bb1 = (const float*)d_in[12];
  const float* Ws2 = (const float*)d_in[13]; const float* gs2 = (const float*)d_in[14]; const float* bs2 = (const float*)d_in[15];
  const float* Wa2 = (const float*)d_in[16]; const float* ga2 = (const float*)d_in[17]; const float* ba2 = (const float*)d_in[18];
  const float* Wb2 = (const float*)d_in[19]; const float* gb2 = (const float*)d_in[20]; const float* bb2 = (const float*)d_in[21];
  const float* Wd2 = (const float*)d_in[22]; const float* gd2 = (const float*)d_in[23]; const float* bd2 = (const float*)d_in[24];
  const float* Ws3 = (const float*)d_in[25]; const float* gs3 = (const float*)d_in[26]; const float* bs3 = (const float*)d_in[27];
  const float* Wa3 = (const float*)d_in[28]; const float* ga3 = (const float*)d_in[29]; const float* ba3 = (const float*)d_in[30];
  const float* Wb3 = (const float*)d_in[31]; const float* gb3 = (const float*)d_in[32]; const float* bb3 = (const float*)d_in[33];
  const float* Wd3 = (const float*)d_in[34]; const float* gd3 = (const float*)d_in[35]; const float* bd3 = (const float*)d_in[36];
  const int* im0  = (const int*)d_in[40]; const int* om0  = (const int*)d_in[41];
  const int* ims1 = (const int*)d_in[42]; const int* oms1 = (const int*)d_in[43];
  const int* imb1 = (const int*)d_in[44]; const int* omb1 = (const int*)d_in[45];
  const int* ims2 = (const int*)d_in[46]; const int* oms2 = (const int*)d_in[47];
  const int* imb2 = (const int*)d_in[48]; const int* omb2 = (const int*)d_in[49];
  const int* ims3 = (const int*)d_in[50]; const int* oms3 = (const int*)d_in[51];
  const int* imb3 = (const int*)d_in[52]; const int* omb3 = (const int*)d_in[53];

  const int n0 = in_sizes[0];
  const int n1 = in_sizes[37] / 3, n2 = in_sizes[38] / 3, n3 = in_sizes[39] / 3;
  const int P0  = in_sizes[40] / 125;
  const int Ps1 = in_sizes[42] / 8,  Pb1 = in_sizes[44] / 27;
  const int Ps2 = in_sizes[46] / 8,  Pb2 = in_sizes[48] / 27;
  const int Ps3 = in_sizes[50] / 8,  Pb3 = in_sizes[52] / 27;

  // workspace: 4 rotating f32 slots + 1 bf16 activation slot + bf16 weights
  long slot = (long)(n0 + 1) * 64;
  long t;
  t = (long)(n1 + 1) * 64;  if (t > slot) slot = t;
  t = (long)(n2 + 1) * 128; if (t > slot) slot = t;
  t = (long)(n3 + 1) * 256; if (t > slot) slot = t;
  slot = (slot + 63) & ~63L;                               // keep 256B alignment
  float* S0 = (float*)d_ws;
  float* S1 = S0 + slot;  float* S2 = S1 + slot;  float* S3 = S2 + slot;
  float* stats = S3 + slot;                                // 512 floats
  bf16_t* XB = (bf16_t*)(stats + 512);                     // slot bf16 elements
  bf16_t* wcur = XB + slot;

  // bf16 transposed weights [K][Cout][Cin]
  bf16_t* ts1 = xpose(Ws1, 0, 8,  64,  64,  &wcur, stream);
  bf16_t* ta1 = xpose(Wa1, 0, 27, 64,  64,  &wcur, stream);
  bf16_t* tb1 = xpose(Wb1, 0, 27, 64,  64,  &wcur, stream);
  bf16_t* ts2 = xpose(Ws2, 0, 8,  64,  64,  &wcur, stream);
  bf16_t* ta2 = xpose(Wa2, 0, 27, 64,  128, &wcur, stream);
  bf16_t* tb2 = xpose(Wb2, 0, 27, 128, 128, &wcur, stream);
  bf16_t* td2 = xpose(Wd2, 0, 1,  64,  128, &wcur, stream);
  bf16_t* ts3 = xpose(Ws3, 0, 8,  128, 128, &wcur, stream);
  bf16_t* ta3 = xpose(Wa3, 0, 27, 128, 256, &wcur, stream);
  bf16_t* tb3 = xpose(Wb3, 0, 27, 256, 256, &wcur, stream);
  bf16_t* td3 = xpose(Wd3, 0, 1,  128, 256, &wcur, stream);

  float* out = (float*)d_out;
  float* y1o = out;
  float* y2o = out + (long)n1 * 64;
  float* y3o = y2o + (long)n2 * 128;

  // ---- stem: 5x5x5 conv (1->64) + BN + ReLU, coords c0 -> c0
  k_zero<<<dim3(1024), 256, 0, stream>>>(S0, (long)(n0 + 1) * 64);
  k_stem<<<dim3((P0 + 3) / 4, 125), 256, 0, stream>>>(feats, W0, im0, om0, S0, n0, P0);
  run_bn(S0, stats, g0, b0, XB, n0, 64, 1, stream);

  // ---- level 1
  run_conv(XB, ts1, ims1, oms1, S1, 8, Ps1, 64, 64, n0, n1, stream);   // stride-2
  run_bn(S1, stats, gs1, bs1, XB, n1, 64, 1, stream);
  run_conv(XB, ta1, imb1, omb1, S2, 27, Pb1, 64, 64, n1, n1, stream);
  run_bn(S2, stats, ga1, ba1, XB, n1, 64, 1, stream);
  run_conv(XB, tb1, imb1, omb1, S3, 27, Pb1, 64, 64, n1, n1, stream);
  run_bn(S3, stats, gb1, bb1, nullptr, n1, 64, 0, stream);
  k_add_relu<<<dim3((unsigned)(((long)(n1 + 1) * 64 + 255) / 256)), 256, 0, stream>>>(
      S3, S1, XB, y1o, n1, 64);                                        // y1

  // ---- level 2
  run_conv(XB, ts2, ims2, oms2, S0, 8, Ps2, 64, 64, n1, n2, stream);
  run_bn(S0, stats, gs2, bs2, XB, n2, 64, 1, stream);
  run_gemm(XB, td2, S1, n2, 64, 128, stream);                          // residual 1x1
  run_conv(XB, ta2, imb2, omb2, S2, 27, Pb2, 64, 128, n2, n2, stream);
  run_bn(S1, stats, gd2, bd2, nullptr, n2, 128, 0, stream);
  run_bn(S2, stats, ga2, ba2, XB, n2, 128, 1, stream);
  run_conv(XB, tb2, imb2, omb2, S3, 27, Pb2, 128, 128, n2, n2, stream);
  run_bn(S3, stats, gb2, bb2, nullptr, n2, 128, 0, stream);
  k_add_relu<<<dim3((unsigned)(((long)(n2 + 1) * 128 + 255) / 256)), 256, 0, stream>>>(
      S3, S1, XB, y2o, n2, 128);                                       // y2

  // ---- level 3
  run_conv(XB, ts3, ims3, oms3, S0, 8, Ps3, 128, 128, n2, n3, stream);
  run_bn(S0, stats, gs3, bs3, XB, n3, 128, 1, stream);
  run_gemm(XB, td3, S1, n3, 128, 256, stream);
  run_conv(XB, ta3, imb3, omb3, S2, 27, Pb3, 128, 256, n3, n3, stream);
  run_bn(S1, stats, gd3, bd3, nullptr, n3, 256, 0, stream);
  run_bn(S2, stats, ga3, ba3, XB, n3, 256, 1, stream);
  run_conv(XB, tb3, imb3, omb3, S3, 27, Pb3, 256, 256, n3, n3, stream);
  run_bn(S3, stats, gb3, bb3, nullptr, n3, 256, 0, stream);
  k_add_relu<<<dim3((unsigned)(((long)(n3 + 1) * 256 + 255) / 256)), 256, 0, stream>>>(
      S3, S1, nullptr, y3o, n3, 256);                                  // y3
}
